// MultiHeadAttention_63445256896561
// MI455X (gfx1250) — compile-verified, matrix operand order checked by reading, not run on hardware
//
#include <hip/hip_runtime.h>
#include <hip/hip_bf16.h>

typedef __attribute__((ext_vector_type(8)))  int          v8i;
typedef __attribute__((ext_vector_type(16))) _Float16     v16h;
typedef __attribute__((ext_vector_type(8)))  float        v8f;
typedef __attribute__((ext_vector_type(4)))  unsigned int u32x4;
typedef __attribute__((ext_vector_type(4)))  int          i32x4;
typedef __attribute__((ext_vector_type(8)))  int          i32x8;

#define DIM_   2048
#define NHEADS 16
#define HDIM   128
#define SEQ    2048
#define BATCH  2
#define MROWS  (BATCH*SEQ)
#define LN_EPS 1e-5f
#define THRESH 0.7f

__device__ __forceinline__ int kmap8(int v, int hi) {
  return ((v & 1) << 2) | ((v & 2) << 3) | ((v >> 2) << 5) | (hi ? 8 : 0);
}
__device__ __forceinline__ int kmap16(int p, int hi) {
  int v = p >> 1;
  return ((((v & 3) << 1) | ((v >> 2) << 4)) | (hi ? 8 : 0)) + (p & 1);
}

__device__ __forceinline__ void tdm_load_tile_2d(unsigned int lds_addr,
                                                 const void* gptr,
                                                 unsigned int tile_rows,
                                                 unsigned int tile_cols_bytes,
                                                 unsigned int row_stride_bytes,
                                                 unsigned int tensor_rows) {
  unsigned long long ga = (unsigned long long)(size_t)gptr;
  u32x4 g0;
  g0[0] = 1u;
  g0[1] = lds_addr;
  g0[2] = (unsigned int)(ga & 0xFFFFFFFFu);
  g0[3] = (unsigned int)((ga >> 32) & 0x01FFFFFFu) | (2u << 30);
  i32x8 g1;
  g1[0] = 0;
  g1[1] = (int)((row_stride_bytes & 0xFFFFu) << 16);
  g1[2] = (int)(((row_stride_bytes >> 16) & 0xFFFFu) |
                ((tensor_rows & 0xFFFFu) << 16));
  g1[3] = (int)(((tensor_rows >> 16) & 0xFFFFu) |
                ((tile_cols_bytes & 0xFFFFu) << 16));
  g1[4] = (int)(tile_rows & 0xFFFFu);
  g1[5] = (int)row_stride_bytes;
  g1[6] = 0;
  g1[7] = 0;
  i32x4 g2 = {0, 0, 0, 0};
  i32x4 g3 = {0, 0, 0, 0};
#if defined(__clang_major__) && (__clang_major__ >= 23)
  i32x8 gx = {0, 0, 0, 0, 0, 0, 0, 0};
  __builtin_amdgcn_tensor_load_to_lds(g0, g1, g2, g3, gx, 0);
#else
  __builtin_amdgcn_tensor_load_to_lds(g0, g1, g2, g3, 0);
#endif
}

__device__ __forceinline__ void async_load_b128(unsigned int lds_addr,
                                                unsigned long long gaddr) {
  asm volatile("global_load_async_to_lds_b128 %0, %1, off"
               :: "v"(lds_addr), "v"(gaddr) : "memory");
}
__device__ __forceinline__ void wait_asynccnt0() {
  asm volatile("s_wait_asynccnt 0" ::: "memory");
}

__device__ __forceinline__ float blockReduceSum(float v) {
  __shared__ float sh[256];
  int tid = threadIdx.x;
  sh[tid] = v; __syncthreads();
  for (int s = 128; s > 0; s >>= 1) {
    if (tid < s) sh[tid] += sh[tid + s];
    __syncthreads();
  }
  float r = sh[0]; __syncthreads();
  return r;
}
__device__ __forceinline__ float blockReduceMax(float v) {
  __shared__ float sh[256];
  int tid = threadIdx.x;
  sh[tid] = v; __syncthreads();
  for (int s = 128; s > 0; s >>= 1) {
    if (tid < s) sh[tid] = fmaxf(sh[tid], sh[tid + s]);
    __syncthreads();
  }
  float r = sh[0]; __syncthreads();
  return r;
}

__global__ void init_stats_kernel(float* stats) {
  if (threadIdx.x < 16) stats[threadIdx.x] = 0.0f;
}

__global__ __launch_bounds__(256) void ln_rowstats_kernel(const float* __restrict__ X,
                                                          float* __restrict__ RS) {
  int row = blockIdx.x;
  const float* x = X + (size_t)row * DIM_;
  float s = 0.f, s2 = 0.f;
  for (int i = threadIdx.x; i < DIM_; i += 256) { float v = x[i]; s += v; s2 += v * v; }
  float ts  = blockReduceSum(s);
  float ts2 = blockReduceSum(s2);
  if (threadIdx.x == 0) {
    float mu  = ts / (float)DIM_;
    float var = fmaxf(ts2 / (float)DIM_ - mu * mu, 0.f);
    RS[2 * row]     = mu;
    RS[2 * row + 1] = rsqrtf(var + LN_EPS);
  }
}

__global__ __launch_bounds__(256) void ln_absmax_kernel(const float* __restrict__ X,
                                                        const float* __restrict__ RS,
                                                        const float* __restrict__ g,
                                                        const float* __restrict__ bta,
                                                        float* slot) {
  size_t total = (size_t)MROWS * DIM_;
  float am = 0.f;
  for (size_t e = (size_t)blockIdx.x * 256 + threadIdx.x; e < total;
       e += (size_t)gridDim.x * 256) {
    int row = (int)(e / DIM_), col = (int)(e % DIM_);
    float v = (X[e] - RS[2 * row]) * RS[2 * row + 1] * g[col] + bta[col];
    am = fmaxf(am, fabsf(v));
  }
  am = blockReduceMax(am);
  if (threadIdx.x == 0) atomicMax((unsigned int*)slot, __float_as_uint(am));
}

__global__ __launch_bounds__(256) void ln_quant_kernel(const float* __restrict__ X,
                                                       const float* __restrict__ RS,
                                                       const float* __restrict__ g,
                                                       const float* __restrict__ bta,
                                                       const float* __restrict__ slot,
                                                       signed char* __restrict__ Xq) {
  size_t total = (size_t)MROWS * DIM_;
  float s = 127.f / fmaxf(slot[0], 1e-5f);
  for (size_t e = (size_t)blockIdx.x * 256 + threadIdx.x; e < total;
       e += (size_t)gridDim.x * 256) {
    int row = (int)(e / DIM_), col = (int)(e % DIM_);
    float v = (X[e] - RS[2 * row]) * RS[2 * row + 1] * g[col] + bta[col];
    float q = fminf(fmaxf(rintf(v * s), -128.f), 127.f);
    Xq[e] = (signed char)q;
  }
}

__global__ __launch_bounds__(256) void w_absum_kernel(const float* __restrict__ W, float* slot) {
  size_t total = (size_t)DIM_ * DIM_;
  float s = 0.f;
  for (size_t e = (size_t)blockIdx.x * 256 + threadIdx.x; e < total;
       e += (size_t)gridDim.x * 256)
    s += fabsf(W[e]);
  s = blockReduceSum(s);
  if (threadIdx.x == 0) atomicAdd(slot, s);
}

__global__ __launch_bounds__(256) void w_mask_kernel(const float* __restrict__ W,
                                                     const float* __restrict__ sumabs,
                                                     float* cntslot, float* msumslot) {
  size_t total = (size_t)DIM_ * DIM_;
  float delta = THRESH * (sumabs[0] / (float)total);
  float c = 0.f, ms = 0.f;
  for (size_t e = (size_t)blockIdx.x * 256 + threadIdx.x; e < total;
       e += (size_t)gridDim.x * 256) {
    float a = fabsf(W[e]);
    if (a > delta) { c += 1.f; ms += a; }
  }
  c  = blockReduceSum(c);
  ms = blockReduceSum(ms);
  if (threadIdx.x == 0) { atomicAdd(cntslot, c); atomicAdd(msumslot, ms); }
}

__global__ __launch_bounds__(256) void w_pack_kernel(const float* __restrict__ W,
                                                     const float* __restrict__ sumabs,
                                                     signed char* __restrict__ Wq) {
  size_t total = (size_t)DIM_ * DIM_;
  float delta = THRESH * (sumabs[0] / (float)total);
  for (size_t e = (size_t)blockIdx.x * 256 + threadIdx.x; e < total;
       e += (size_t)gridDim.x * 256) {
    float w = W[e];
    Wq[e] = (fabsf(w) > delta) ? (signed char)(w > 0.f ? 1 : -1) : (signed char)0;
  }
}

template <typename OutT>
__global__ __launch_bounds__(256) void gemm_i8_kernel(const signed char* __restrict__ Xq,
                                                      const signed char* __restrict__ Wq,
                                                      OutT* __restrict__ Out,
                                                      const float* __restrict__ stats,
                                                      int proj) {
  __shared__ signed char Alds[64][64];
  __shared__ signed char Blds[128][64];
  const int Ktot = DIM_, Ntot = DIM_;
  int tid = threadIdx.x, lane = tid & 31, wave = tid >> 5;
  int hi = lane >> 4, mr = lane & 15;
  int wm = (wave >> 1) * 16, wn = (wave & 1) * 64;
  int bm = blockIdx.y * 64, bn = blockIdx.x * 128;

  unsigned int aoff = (unsigned int)(size_t)(void*)&Alds[0][0];
  unsigned int boff = (unsigned int)(size_t)(void*)&Blds[0][0];

  v8i acc[4];
#pragma unroll
  for (int t = 0; t < 4; ++t)
#pragma unroll
    for (int i = 0; i < 8; ++i) acc[t][i] = 0;

  for (int k0 = 0; k0 < Ktot; k0 += 64) {
    __syncthreads();
    if (wave == 0) {
      tdm_load_tile_2d(aoff, Xq + (size_t)bm * Ktot + k0, 64, 64, Ktot, MROWS);
      tdm_load_tile_2d(boff, Wq + (size_t)bn * Ktot + k0, 128, 64, Ktot, DIM_);
      __builtin_amdgcn_s_wait_tensorcnt(0);
    }
    if (k0 + 64 < Ktot) {
      __builtin_prefetch(&Xq[(size_t)(bm + (tid >> 2)) * Ktot + k0 + 64], 0, 1);
      __builtin_prefetch(&Wq[(size_t)(bn + (tid >> 1)) * Ktot + k0 + 64], 0, 1);
    }
    __syncthreads();
    v8i a;
#pragma unroll
    for (int v = 0; v < 8; ++v)
      a[v] = *(const int*)&Alds[wm + mr][kmap8(v, hi)];
#pragma unroll
    for (int t = 0; t < 4; ++t) {
      v8i b;
#pragma unroll
      for (int v = 0; v < 8; ++v)
        b[v] = *(const int*)&Blds[wn + t * 16 + mr][kmap8(v, hi)];
      acc[t] = __builtin_amdgcn_wmma_i32_16x16x64_iu8(true, a, true, b, acc[t], false, false);
    }
  }

  float alpha = stats[12 + proj] / fmaxf(stats[8 + proj], 1.f);
  float invs  = fmaxf(stats[proj], 1e-5f) / 127.f;
  float scale = alpha * invs;
#pragma unroll
  for (int t = 0; t < 4; ++t)
#pragma unroll
    for (int r = 0; r < 8; ++r) {
      int row = bm + wm + r + hi * 8;
      int col = bn + wn + t * 16 + mr;
      Out[(size_t)row * Ntot + col] = (OutT)(scale * (float)acc[t][r]);
    }
}

__global__ __launch_bounds__(256) void rope_kernel(_Float16* __restrict__ X) {
  size_t np = (size_t)MROWS * DIM_ / 2;
  for (size_t i = (size_t)blockIdx.x * 256 + threadIdx.x; i < np;
       i += (size_t)gridDim.x * 256) {
    size_t row = i / (DIM_ / 2);
    int d = (int)(i % (DIM_ / 2)) * 2;
    int h = d >> 7;
    int j = (d & 127) >> 1;
    float inv = __expf(-(float)(2 * j) * (1.0f / (float)HDIM) * 9.210340371976184f);
    float ang = (float)h * inv;
    float c = __cosf(ang), s = __sinf(ang);
    size_t base = row * DIM_ + d;
    float a = (float)X[base], b = (float)X[base + 1];
    X[base]     = (_Float16)(a * c - b * s);
    X[base + 1] = (_Float16)(a * s + b * c);
  }
}

__global__ __launch_bounds__(128) void flash_attn_kernel(const _Float16* __restrict__ Q,
                                                         const _Float16* __restrict__ K,
                                                         const _Float16* __restrict__ V,
                                                         float* __restrict__ Att) {
  __shared__ _Float16 Kt[32][HDIM];
  __shared__ _Float16 Vt[32][HDIM];
  __shared__ _Float16 Pl[4][16][32];

  int bh = blockIdx.y;
  int b = bh >> 4, h = bh & 15;
  int qblk = blockIdx.x;
  int tid = threadIdx.x, lane = tid & 31, wave = tid >> 5;
  int hi = lane >> 4, n = lane & 15;
  const float scale = 0.08838834764831845f;

  unsigned int kt_base = (unsigned int)(size_t)(void*)&Kt[0][0];
  unsigned int vt_base = (unsigned int)(size_t)(void*)&Vt[0][0];

  int qrow = qblk * 64 + wave * 16 + n;
  const _Float16* qbase = Q + ((size_t)(b * SEQ + qrow)) * DIM_ + h * HDIM;

  v16h qf[4];
#pragma unroll
  for (int kc = 0; kc < 4; ++kc)
#pragma unroll
    for (int p = 0; p < 16; ++p) {
      int d = kc * 32 + kmap16(p, hi);
      qf[kc][p] = (_Float16)((float)qbase[d] * scale);
    }

  v8f out[8];
#pragma unroll
  for (int f = 0; f < 8; ++f)
#pragma unroll
    for (int r = 0; r < 8; ++r) out[f][r] = 0.f;
  float mst[8], lst[8];
#pragma unroll
  for (int r = 0; r < 8; ++r) { mst[r] = -1e30f; lst[r] = 0.f; }

  for (int kb = 0; kb < SEQ / 32; ++kb) {
    __syncthreads();
#pragma unroll
    for (int i = 0; i < 4; ++i) {
      int r = wave * 8 + i * 2 + (lane >> 4);
      int ch = (lane & 15) * 8;
      size_t gidx = ((size_t)(b * SEQ + kb * 32 + r)) * DIM_ + h * HDIM + ch;
      unsigned int loff = (unsigned int)(r * (HDIM * 2) + ch * 2);
      async_load_b128(kt_base + loff, (unsigned long long)(size_t)&K[gidx]);
      async_load_b128(vt_base + loff, (unsigned long long)(size_t)&V[gidx]);
    }
    wait_asynccnt0();
    __syncthreads();

    v8f s0, s1;
#pragma unroll
    for (int r = 0; r < 8; ++r) { s0[r] = 0.f; s1[r] = 0.f; }
#pragma unroll
    for (int kc = 0; kc < 4; ++kc) {
      v16h b0, b1;
#pragma unroll
      for (int p = 0; p < 16; ++p) {
        int dk = kc * 32 + kmap16(p, hi);
        b0[p] = Kt[n][dk];
        b1[p] = Kt[16 + n][dk];
      }
      s0 = __builtin_amdgcn_wmma_f32_16x16x32_f16(false, qf[kc], false, b0, (short)0, s0, false, false);
      s1 = __builtin_amdgcn_wmma_f32_16x16x32_f16(false, qf[kc], false, b1, (short)0, s1, false, false);
    }

    float pcorr[8];
#pragma unroll
    for (int r = 0; r < 8; ++r) {
      float v0 = s0[r], v1 = s1[r];
      float mx = fmaxf(v0, v1);
#pragma unroll
      for (int off = 1; off < 16; off <<= 1) mx = fmaxf(mx, __shfl_xor(mx, off, 32));
      float nm = fmaxf(mst[r], mx);
      float corr = __expf(mst[r] - nm);
      float p0 = __expf(v0 - nm), p1 = __expf(v1 - nm);
      float rs = p0 + p1;
#pragma unroll
      for (int off = 1; off < 16; off <<= 1) rs += __shfl_xor(rs, off, 32);
      lst[r] = lst[r] * corr + rs;
      mst[r] = nm;
      pcorr[r] = corr;
      int row = r + hi * 8;
      Pl[wave][row][n]      = (_Float16)p0;
      Pl[wave][row][n + 16] = (_Float16)p1;
    }
#pragma unroll
    for (int f = 0; f < 8; ++f)
#pragma unroll
      for (int r = 0; r < 8; ++r) out[f][r] *= pcorr[r];

    v16h pa;
#pragma unroll
    for (int p = 0; p < 16; ++p) pa[p] = Pl[wave][n][kmap16(p, hi)];
#pragma unroll
    for (int f = 0; f < 8; ++f) {
      v16h vb;
#pragma unroll
      for (int p = 0; p < 16; ++p) {
        int t = kmap16(p, hi);
        vb[p] = Vt[t][f * 16 + n];
      }
      out[f] = __builtin_amdgcn_wmma_f32_16x16x32_f16(false, pa, false, vb, (short)0, out[f], false, false);
    }
  }

#pragma unroll
  for (int f = 0; f < 8; ++f)
#pragma unroll
    for (int r = 0; r < 8; ++r) {
      int row = qblk * 64 + wave * 16 + r + hi * 8;
      float invl = 1.0f / fmaxf(lst[r], 1e-20f);
      Att[((size_t)(b * SEQ + row)) * DIM_ + h * HDIM + f * 16 + n] = out[f][r] * invl;
    }
}

extern "C" void kernel_launch(void* const* d_in, const int* in_sizes, int n_in,
                              void* d_out, int out_size, void* d_ws, size_t ws_size,
                              hipStream_t stream) {
  (void)in_sizes; (void)n_in; (void)out_size; (void)ws_size;
  const float* x    = (const float*)d_in[0];
  const float* w[4]  = {(const float*)d_in[1], (const float*)d_in[4],
                        (const float*)d_in[7], (const float*)d_in[10]};
  const float* g[4]  = {(const float*)d_in[2], (const float*)d_in[5],
                        (const float*)d_in[8], (const float*)d_in[11]};
  const float* bb[4] = {(const float*)d_in[3], (const float*)d_in[6],
                        (const float*)d_in[9], (const float*)d_in[12]};

  char* ws = (char*)d_ws;
  float* stats = (float*)ws;
  size_t off = 256;
  float* rs_x = (float*)(ws + off); off += (size_t)2 * MROWS * 4;
  float* rs_o = (float*)(ws + off); off += (size_t)2 * MROWS * 4;
  off = (off + 255) & ~(size_t)255;
  signed char* xq[4];
  for (int p = 0; p < 4; ++p) { xq[p] = (signed char*)(ws + off); off += (size_t)MROWS * DIM_; }
  signed char* wq8[4];
  for (int p = 0; p < 4; ++p) { wq8[p] = (signed char*)(ws + off); off += (size_t)DIM_ * DIM_; }
  _Float16* qkvh[3];
  for (int p = 0; p < 3; ++p) { qkvh[p] = (_Float16*)(ws + off); off += (size_t)MROWS * DIM_ * 2; }
  float* att = (float*)(ws + off); off += (size_t)MROWS * DIM_ * 4;

  init_stats_kernel<<<1, 64, 0, stream>>>(stats);
  ln_rowstats_kernel<<<MROWS, 256, 0, stream>>>(x, rs_x);
  for (int p = 0; p < 3; ++p)
    ln_absmax_kernel<<<1024, 256, 0, stream>>>(x, rs_x, g[p], bb[p], &stats[p]);
  for (int p = 0; p < 3; ++p)
    ln_quant_kernel<<<2048, 256, 0, stream>>>(x, rs_x, g[p], bb[p], &stats[p], xq[p]);
  for (int p = 0; p < 4; ++p) {
    w_absum_kernel<<<1024, 256, 0, stream>>>(w[p], &stats[4 + p]);
    w_mask_kernel<<<1024, 256, 0, stream>>>(w[p], &stats[4 + p], &stats[8 + p], &stats[12 + p]);
    w_pack_kernel<<<2048, 256, 0, stream>>>(w[p], &stats[4 + p], wq8[p]);
  }

  dim3 ggrid(DIM_ / 128, MROWS / 64);
  for (int p = 0; p < 3; ++p)
    gemm_i8_kernel<_Float16><<<ggrid, 256, 0, stream>>>(xq[p], wq8[p], qkvh[p], stats, p);

  rope_kernel<<<2048, 256, 0, stream>>>(qkvh[0]);
  rope_kernel<<<2048, 256, 0, stream>>>(qkvh[1]);

  flash_attn_kernel<<<dim3(SEQ / 64, BATCH * NHEADS), 128, 0, stream>>>(
      qkvh[0], qkvh[1], qkvh[2], att);

  ln_rowstats_kernel<<<MROWS, 256, 0, stream>>>(att, rs_o);
  ln_absmax_kernel<<<1024, 256, 0, stream>>>(att, rs_o, g[3], bb[3], &stats[3]);
  ln_quant_kernel<<<2048, 256, 0, stream>>>(att, rs_o, g[3], bb[3], &stats[3], xq[3]);
  gemm_i8_kernel<float><<<ggrid, 256, 0, stream>>>(xq[3], wq8[3], (float*)d_out, stats, 3);
}
